// VariableShapeAttention_11312943857928
// MI455X (gfx1250) — matrix operand order repaired and checked by reading, not compiled, not run on hardware
//
#include <hip/hip_runtime.h>

typedef __attribute__((ext_vector_type(16))) _Float16 v16h;
typedef __attribute__((ext_vector_type(8)))  _Float16 v8h;
typedef __attribute__((ext_vector_type(4)))  _Float16 v4h;
typedef __attribute__((ext_vector_type(2)))  _Float16 v2h;
typedef __attribute__((ext_vector_type(8)))  float    v8f;
typedef __attribute__((ext_vector_type(4)))  float    v4f;

namespace {

constexpr int   kNH       = 12;
constexpr int   kKsStride = 40;   // halves (80 B) per K row: 16B aligned, bank-friendly
constexpr int   kVtStride = 520;  // halves (1040 B) per Vt row
constexpr int   kPbStride = 40;   // halves per P-buffer row (32 keys + pad)
constexpr float kLog2e    = 1.4426950408889634f;
constexpr float kScaleL2  = 0.17677669529663687f * kLog2e;  // (32^-0.5) * log2(e)
constexpr float kMaskL2   = -100.0f * kLog2e;

__device__ __forceinline__ int regof(int g) { return g < 24 ? 0 : (g < 28 ? 1 : 2); }

__device__ __forceinline__ v16h cat8(v8h lo, v8h hi) {
  return __builtin_shufflevector(lo, hi, 0, 1, 2, 3, 4, 5, 6, 7,
                                 8, 9, 10, 11, 12, 13, 14, 15);
}

__device__ __forceinline__ v8h pack8(float p0, float p1, float p2, float p3,
                                     float p4, float p5, float p6, float p7) {
  v2h a0 = __builtin_bit_cast(v2h, __builtin_amdgcn_cvt_pkrtz(p0, p1));
  v2h a1 = __builtin_bit_cast(v2h, __builtin_amdgcn_cvt_pkrtz(p2, p3));
  v2h a2 = __builtin_bit_cast(v2h, __builtin_amdgcn_cvt_pkrtz(p4, p5));
  v2h a3 = __builtin_bit_cast(v2h, __builtin_amdgcn_cvt_pkrtz(p6, p7));
  v4h c0 = __builtin_shufflevector(a0, a1, 0, 1, 2, 3);
  v4h c1 = __builtin_shufflevector(a2, a3, 0, 1, 2, 3);
  return __builtin_shufflevector(c0, c1, 0, 1, 2, 3, 4, 5, 6, 7);
}

// ---- bias pre-pass: bias_t[h][qt][kt][lane][r] (f32, pre-scaled by log2e) ----
// Transposed score fragment: lane%16 = query col, r + 8*(lane>=16) = key row.
__global__ __launch_bounds__(256) void swin_bias_kernel(
    const float* __restrict__ bias_table, float* __restrict__ bias_t) {
  int idx  = blockIdx.x * 256 + threadIdx.x;  // 12*32*32*256 total
  int r    = idx & 7;
  int lane = (idx >> 3) & 31;
  int kt   = (idx >> 8) & 31;
  int qt   = (idx >> 13) & 31;
  int h    = idx >> 18;
  int qn = qt * 16 + (lane & 15);             // query = N col = lane%16
  int kn = kt * 16 + r + ((lane >> 4) << 3);  // key   = M row = r + 8*(lane>=16)
  int qd = qn >> 6, qh = (qn >> 3) & 7, qw = qn & 7;
  int kd = kn >> 6, kh = (kn >> 3) & 7, kw = kn & 7;
  int rel = (qd - kd + 7) * 225 + (qh - kh + 7) * 15 + (qw - kw + 7);
  bias_t[idx] = bias_table[rel * kNH + h] * kLog2e;
}

// ---- core flash-attention loop for one (window, head); HASMASK = boundary window ----
template <bool HASMASK>
__device__ __forceinline__ void attn_core(
    const float* __restrict__ qkv, const float* __restrict__ bias_t,
    float* __restrict__ out, const _Float16* Ks, const _Float16* Vt, _Float16* myP,
    int h, int bd, int bh, int bw, int lane, int wave) {
  const int col = lane & 15;
  const int grp = lane >> 4;
  const int kbA = grp ? 8 : 0;    // A-fragment K-chunk base (interleaved layout)
  const int kbB = grp ? 16 : 0;   // B-fragment K base (contiguous layout)

  const v16h ones = {(_Float16)1.f, (_Float16)1.f, (_Float16)1.f, (_Float16)1.f,
                     (_Float16)1.f, (_Float16)1.f, (_Float16)1.f, (_Float16)1.f,
                     (_Float16)1.f, (_Float16)1.f, (_Float16)1.f, (_Float16)1.f,
                     (_Float16)1.f, (_Float16)1.f, (_Float16)1.f, (_Float16)1.f};

  // key w-coord region codes (key w-coord within window == r)
  int kwreg[8];
  if (HASMASK) {
#pragma unroll
    for (int r = 0; r < 8; ++r) kwreg[r] = regof(bw * 8 + r);
  }

  for (int qt = wave; qt < 32; qt += 8) {
    const int qbase = qt * 16;
    const int qn = qbase + col;
    const int zqw = qn & 7, zqh = (qn >> 3) & 7, zqd = qn >> 6;
    const long qsp = (((long)((bd * 8 + zqd + 4) & 31) * 32 + ((bh * 8 + zqh + 4) & 31)) * 32 +
                      ((bw * 8 + zqw + 4) & 31));

    // Q B-fragment (contiguous hd chunk) loaded directly from global, once per tile.
    v16h bq;
    {
      long qoff = qsp * 1152 + h;
#pragma unroll
      for (int j = 0; j < 16; ++j)
        bq[j] = (_Float16)qkv[qoff + (kbB + j) * 12];  // hd = kbB..kbB+15
    }

    int qreg = 0;
    if (HASMASK)
      qreg = regof(bd * 8 + zqd) * 9 + regof(bh * 8 + zqh) * 3 + regof(bw * 8 + zqw);

    v8f O0 = {}, O1 = {};           // O' = V^T x P' : lane = query, r+8*grp = head-dim
    float mrun = -1e30f, lrun = 0.0f;

    const float* biasQ = bias_t + (((long)h * 32 + qt) * 32) * 256 + lane * 8;

    for (int blk = 0; blk < 16; ++blk) {   // 32 keys per block
      float sv[16];
#pragma unroll
      for (int sub = 0; sub < 2; ++sub) {
        const int kt = blk * 2 + sub;
        const int krow = kt * 16 + col;    // A-fragment: lane%16 = key row

        const _Float16* ka = Ks + krow * kKsStride;
        v16h aK = cat8(*(const v8h*)(ka + kbB), *(const v8h*)(ka + kbB + 8));

        v8f zero = {};
        v8f s = __builtin_amdgcn_wmma_f32_16x16x32_f16(false, aK, false, bq,
                                                       (short)0, zero, false, false);
        const float* bt = biasQ + kt * 256;
        v4f b0 = *(const v4f*)bt;
        v4f b1 = *(const v4f*)(bt + 4);

        int kbase2 = 0;
        if (HASMASK) {
          int zdd = blk >> 1;                       // key d-coord
          int zhh = (blk * 4 + sub * 2 + grp) & 7;  // key h-coord
          kbase2 = regof(bd * 8 + zdd) * 9 + regof(bh * 8 + zhh) * 3;
        }
#pragma unroll
        for (int r = 0; r < 8; ++r) {
          float bsv = (r < 4) ? b0[r] : b1[r - 4];
          float v = s[r] * kScaleL2 + bsv;                 // log2-domain logits
          if (HASMASK) v += (qreg == kbase2 + kwreg[r]) ? 0.0f : kMaskL2;
          sv[sub * 8 + r] = v;
        }
      }

      // ---- per-query max: in-lane tree + one xor-16 exchange ----
      float mt = sv[0];
#pragma unroll
      for (int i = 1; i < 16; ++i) mt = fmaxf(mt, sv[i]);
      mt = fmaxf(mt, __shfl_xor(mt, 16, 32));
      float mn = fmaxf(mrun, mt);
      float alpha = __builtin_amdgcn_exp2f(mrun - mn);
      mrun = mn;

      // ---- exponentiate + stage P' (f16) into per-wave LDS buffer ----
      float p[16];
#pragma unroll
      for (int i = 0; i < 16; ++i) p[i] = __builtin_amdgcn_exp2f(sv[i] - mn);
      {
        _Float16* prow = myP + col * kPbStride + grp * 8;
        *(v8h*)(prow)      = pack8(p[0], p[1], p[2], p[3], p[4], p[5], p[6], p[7]);
        *(v8h*)(prow + 16) = pack8(p[8], p[9], p[10], p[11], p[12], p[13], p[14], p[15]);
      }
      // intra-wave cross-lane LDS RAW: DS pipe is in-order per wave; wait + fence
      __asm__ volatile("s_wait_dscnt 0" ::: "memory");

      // re-read P' as B fragment (interleaved key chunks per query column)
      const _Float16* prow = myP + col * kPbStride;
      v16h bp = cat8(*(const v8h*)(prow + kbA), *(const v8h*)(prow + kbA + 16));

      // per-query sum of P' via WMMA against all-ones A (every row = column sum)
      v8f zv = {};
      v8f Ls = __builtin_amdgcn_wmma_f32_16x16x32_f16(false, ones, false, bp,
                                                      (short)0, zv, false, false);
      lrun = lrun * alpha + Ls[0];

#pragma unroll
      for (int r = 0; r < 8; ++r) { O0[r] *= alpha; O1[r] *= alpha; }

      // A = V^T tiles (d rows, interleaved key chunks within the block)
      const int koff = blk * 32;
      const _Float16* va0 = Vt + col * kVtStride + koff;         // d = 0..15
      const _Float16* va1 = Vt + (col + 16) * kVtStride + koff;  // d = 16..31
      v16h av0 = cat8(*(const v8h*)(va0 + kbA), *(const v8h*)(va0 + kbA + 16));
      v16h av1 = cat8(*(const v8h*)(va1 + kbA), *(const v8h*)(va1 + kbA + 16));

      O0 = __builtin_amdgcn_wmma_f32_16x16x32_f16(false, av0, false, bp,
                                                  (short)0, O0, false, false);
      O1 = __builtin_amdgcn_wmma_f32_16x16x32_f16(false, av1, false, bp,
                                                  (short)0, O1, false, false);
    }

    // normalize and scatter: lane owns one query token; d = r + 8*grp (+16 for O1)
    float inv = __builtin_amdgcn_rcpf(lrun);
    float* orow = out + qsp * 384 + h;
#pragma unroll
    for (int r = 0; r < 8; ++r) {
      orow[(r + 8 * grp) * 12]      = O0[r] * inv;
      orow[(16 + r + 8 * grp) * 12] = O1[r] * inv;
    }
  }
}

// ---- fused shifted-window attention: one block per (window, head) ----
__global__ __launch_bounds__(256) void swin_attn_kernel(
    const float* __restrict__ qkv, const float* __restrict__ bias_t,
    float* __restrict__ out) {
  __shared__ _Float16 Ks[512 * kKsStride];     // K row-major [token][hd]
  __shared__ _Float16 Vt[32 * kVtStride];      // V transposed [hd][token]
  __shared__ _Float16 Pb[8 * 16 * kPbStride];  // per-wave P' staging [query][key]

  const int h   = blockIdx.x % kNH;
  const int win = blockIdx.x / kNH;
  const int bw = win & 3, bh = (win >> 2) & 3, bd = win >> 4;

  const int tid  = threadIdx.x;
  const int lane = tid & 31;
  const int wave = tid >> 5;

  // ---- stage K and V into LDS as f16 (rolled-window gather) ----
  for (int i = tid; i < 512 * 32; i += 256) {
    int d = i & 31, tok = i >> 5;
    int zw = tok & 7, zh = (tok >> 3) & 7, zd = tok >> 6;
    int sd = (bd * 8 + zd + 4) & 31;
    int sh = (bh * 8 + zh + 4) & 31;
    int sw = (bw * 8 + zw + 4) & 31;
    long base = (((long)sd * 32 + sh) * 32 + sw) * 1152 + d * 12 + h;
    Ks[tok * kKsStride + d] = (_Float16)qkv[base + 384];
    Vt[d * kVtStride + tok] = (_Float16)qkv[base + 768];
  }
  __syncthreads();

  _Float16* myP = Pb + wave * 16 * kPbStride;

  // interior windows (27/64) have an identically-zero shift mask
  if ((bd == 3) || (bh == 3) || (bw == 3)) {
    attn_core<true>(qkv, bias_t, out, Ks, Vt, myP, h, bd, bh, bw, lane, wave);
  } else {
    attn_core<false>(qkv, bias_t, out, Ks, Vt, myP, h, bd, bh, bw, lane, wave);
  }
}

}  // namespace

extern "C" void kernel_launch(void* const* d_in, const int* in_sizes, int n_in,
                              void* d_out, int out_size, void* d_ws, size_t ws_size,
                              hipStream_t stream) {
  const float* qkv        = (const float*)d_in[0];   // [1,32,32,32,1152] f32
  const float* bias_table = (const float*)d_in[1];   // [3375,12] f32
  float* out    = (float*)d_out;                     // [1,32,32,32,384] f32
  float* bias_t = (float*)d_ws;                      // 12*32*32*256 f32 = 12.6 MB

  // 12*32*32 tiles * 256 elements / 256 threads
  swin_bias_kernel<<<12288, 256, 0, stream>>>(bias_table, bias_t);
  // 64 windows * 12 heads
  swin_attn_kernel<<<768, 256, 0, stream>>>(qkv, bias_t, out);
}
